// ScaledDotProductAttention_16088947491512
// MI455X (gfx1250) — matrix operand order repaired and checked by reading, not compiled, not run on hardware
//
#include <hip/hip_runtime.h>

typedef __attribute__((ext_vector_type(16))) _Float16 v16h;
typedef __attribute__((ext_vector_type(8)))  float    v8f;

#define NB 4
#define NH 16
#define NS 1024
#define ND 64
#define SCALE_INV 0.125f   // 1/sqrt(64)

// One workgroup = 4 waves; each wave owns 16 query rows (64 rows per WG).
// Per-wave LDS: 16 x 1024 f32 score row-block (64KB); 4 waves -> 256KB (<320KB WGP LDS).
__global__ __launch_bounds__(128)
void sdpa_wmma_kernel(const float* __restrict__ q,
                      const float* __restrict__ k,
                      const float* __restrict__ v,
                      float* __restrict__ out,
                      float* __restrict__ attn)
{
    extern __shared__ float smem[];
    const int lane  = threadIdx.x & 31;
    const int wave  = threadIdx.x >> 5;
    const int h16   = lane >> 4;     // half-wave selector (0/1)
    const int n     = lane & 15;     // column-within-tile / row-within-tile id
    const int bh    = blockIdx.x >> 4;      // 0..63  (b*16+h)
    const int qblk  = blockIdx.x & 15;      // 0..15
    const int qbase = qblk * 64 + wave * 16;

    const float* Q = q + (size_t)bh * NS * ND;
    const float* K = k + (size_t)bh * NS * ND;
    const float* V = v + (size_t)bh * NS * ND;
    float* Srow = smem + (size_t)wave * 16 * NS;   // this wave's 16x1024 block

    // ---------------- Q fragments: A-matrix 16x32 f16, two d-chunks, pre-scaled
    v16h aq[2];
    #pragma unroll
    for (int c = 0; c < 2; ++c) {
        const float* qrow = Q + (size_t)(qbase + n) * ND + c * 32;
        #pragma unroll
        for (int i = 0; i < 8; ++i) {
            aq[c][i]     = (_Float16)(qrow[     8 * h16 + i] * SCALE_INV);
            aq[c][i + 8] = (_Float16)(qrow[16 + 8 * h16 + i] * SCALE_INV);
        }
    }

    // ---------------- Phase 1: scores S = (Q/8) @ K^T, tile by tile into LDS
    for (int j = 0; j < 64; ++j) {
        if (j < 63)
            __builtin_prefetch(K + (size_t)(16 * (j + 1) + n) * ND, 0, 1);
        v8f acc = {};
        #pragma unroll
        for (int c = 0; c < 2; ++c) {
            // B-matrix 32x16 f16: B[kk,nn] = K[16j+nn, 32c+kk]; kk = 8h+i (i<8), 16+8h+(i-8) (i>=8)
            v16h bk;
            const float* kb = K + (size_t)(16 * j + n) * ND + c * 32;
            #pragma unroll
            for (int i = 0; i < 16; ++i)
                bk[i] = (_Float16)kb[i + 8 * h16 + (i >= 8 ? 8 : 0)];
            acc = __builtin_amdgcn_wmma_f32_16x16x32_f16(
                false, aq[c], false, bk, (short)0, acc, false, false);
        }
        // C layout: lane holds col n, rows r+8*h16 in acc[r]
        #pragma unroll
        for (int r = 0; r < 8; ++r)
            Srow[(size_t)(r + 8 * h16) * NS + 16 * j + n] = acc[r];
    }

    asm volatile("s_wait_dscnt 0" ::: "memory");   // LDS stores visible to all lanes of wave

    // ---------------- Phase 2: row softmax in LDS (exact), write attn to HBM once
    float* attnBase = attn + (size_t)bh * NS * NS;
    for (int r = 0; r < 16; ++r) {
        float* row = Srow + (size_t)r * NS;
        float m = -__builtin_inff();
        for (int t = lane; t < NS; t += 32) m = fmaxf(m, row[t]);
        #pragma unroll
        for (int off = 16; off > 0; off >>= 1) m = fmaxf(m, __shfl_xor(m, off, 32));

        float lsum = 0.0f;
        for (int t = lane; t < NS; t += 32) {
            float p = __expf(row[t] - m);
            row[t] = p;
            lsum += p;
        }
        #pragma unroll
        for (int off = 16; off > 0; off >>= 1) lsum += __shfl_xor(lsum, off, 32);
        const float inv = 1.0f / lsum;

        float* arow = attnBase + (size_t)(qbase + r) * NS;
        for (int t = lane; t < NS; t += 32) {
            float p = row[t] * inv;
            row[t]  = p;        // keep normalized P for phase 3
            arow[t] = p;        // coalesced attn write
        }
    }

    asm volatile("s_wait_dscnt 0" ::: "memory");

    // ---------------- Phase 3: O = P @ V, K-dim (seq) chunks of 32
    v8f o0 = {}, o1 = {}, o2 = {}, o3 = {};
    for (int j2 = 0; j2 < 32; ++j2) {
        // A-matrix 16x32 f16 from normalized P in LDS
        v16h pa;
        const float* prow = Srow + (size_t)n * NS + 32 * j2;
        #pragma unroll
        for (int i = 0; i < 8; ++i) {
            pa[i]     = (_Float16)prow[     8 * h16 + i];
            pa[i + 8] = (_Float16)prow[16 + 8 * h16 + i];
        }
        // B-matrix 32x16 f16 from V: B[kk,nn] = V[32*j2+kk, 16t+nn]; kk = 8h+i (i<8), 16+8h+(i-8) (i>=8)
        const float* vb = V + (size_t)(32 * j2) * ND;
        v16h b0, b1, b2, b3;
        #pragma unroll
        for (int i = 0; i < 16; ++i) {
            const float* vr = vb + (size_t)(i + 8 * h16 + (i >= 8 ? 8 : 0)) * ND;
            b0[i] = (_Float16)vr[n];
            b1[i] = (_Float16)vr[16 + n];
            b2[i] = (_Float16)vr[32 + n];
            b3[i] = (_Float16)vr[48 + n];
        }
        o0 = __builtin_amdgcn_wmma_f32_16x16x32_f16(false, pa, false, b0, (short)0, o0, false, false);
        o1 = __builtin_amdgcn_wmma_f32_16x16x32_f16(false, pa, false, b1, (short)0, o1, false, false);
        o2 = __builtin_amdgcn_wmma_f32_16x16x32_f16(false, pa, false, b2, (short)0, o2, false, false);
        o3 = __builtin_amdgcn_wmma_f32_16x16x32_f16(false, pa, false, b3, (short)0, o3, false, false);
    }

    // ---------------- Write O (C layout -> row-major global)
    float* obase = out + (size_t)bh * NS * ND;
    #pragma unroll
    for (int r = 0; r < 8; ++r) {
        const int m = r + 8 * h16;
        float* dst = obase + (size_t)(qbase + m) * ND;
        dst[     n] = o0[r];
        dst[16 + n] = o1[r];
        dst[32 + n] = o2[r];
        dst[48 + n] = o3[r];
    }
}

extern "C" void kernel_launch(void* const* d_in, const int* in_sizes, int n_in,
                              void* d_out, int out_size, void* d_ws, size_t ws_size,
                              hipStream_t stream) {
    const float* q = (const float*)d_in[0];
    const float* k = (const float*)d_in[1];
    const float* v = (const float*)d_in[2];
    float* out  = (float*)d_out;
    float* attn = out + (size_t)NB * NH * NS * ND;   // tuple: (out, attn) concatenated

    dim3 grid(NB * NH * (NS / 64));   // 1024 workgroups
    dim3 block(128);                  // 4 waves (wave32)
    size_t smem = 4u * 16u * NS * sizeof(float);   // 256 KB dynamic LDS (CDNA5: 320KB/WGP)
    hipLaunchKernelGGL(sdpa_wmma_kernel, grid, block, smem, stream,
                       q, k, v, out, attn);
}
